// GqaAndSwa_18253611008456
// MI455X (gfx1250) — matrix operand order repaired and checked by reading, not compiled, not run on hardware
//
#include <hip/hip_runtime.h>
#include <hip/hip_bf16.h>

// ---------------------------------------------------------------------------
// GQA + sliding-window attention for MI455X (gfx1250, wave32, WMMA).
// Matmul work (4 projections + QK^T + PV) runs on v_wmma_f32_16x16x32_bf16.
// Tile staging uses GLOBAL_LOAD_ASYNC_TO_LDS (ASYNCcnt) with double buffering.
// ---------------------------------------------------------------------------

typedef __bf16 bf16;
typedef __attribute__((ext_vector_type(16))) __bf16 v16bf;
typedef __attribute__((ext_vector_type(8)))  float  v8f;

// GCC-vector int4 pointers in explicit address spaces, matching the builtin's
// signature: (int4 AS1* gsrc, int4 AS3* ldsdst, imm offset, imm cpol)
typedef int v4i __attribute__((vector_size(16)));
typedef __attribute__((address_space(1))) v4i* v4i_gptr;
typedef __attribute__((address_space(3))) v4i* v4i_lptr;

#define B_   2
#define S_   2048
#define D_   2048
#define H_   32
#define HD_  64
#define WIN_ 512
#define NEGF (-1e30f)

// ---------------- async memory->LDS (gfx1250) ------------------------------

__device__ __forceinline__ void async_ld16(void* lds, const void* gptr) {
#if __has_builtin(__builtin_amdgcn_global_load_async_to_lds_b128)
  __builtin_amdgcn_global_load_async_to_lds_b128(
      (v4i_gptr)(void*)gptr, (v4i_lptr)lds, 0, 0);
#else
  *(uint4*)lds = *(const uint4*)gptr;
#endif
}

__device__ __forceinline__ void wait_async() {
#if __has_builtin(__builtin_amdgcn_s_wait_asynccnt)
  __builtin_amdgcn_s_wait_asynccnt(0);
#else
  asm volatile("s_wait_asynccnt 0" ::: "memory");
#endif
}

// ---------------- WMMA helpers (ISA 7.12 register layouts, wave32) ----------

__device__ __forceinline__ v8f wmma_bf16(v16bf a, v16bf b, v8f c) {
  return __builtin_amdgcn_wmma_f32_16x16x32_bf16(false, a, false, b, (short)0, c,
                                                 false, false);
}

// A-matrix 16x32 bf16 fragment from row-major LDS tile (stride in elements).
// Lane L<16: row=L, K={0..7, 16..23}; lane L>=16: row=L-16, K={8..15, 24..31}.
__device__ __forceinline__ v16bf frag_a(const bf16* p, int stride, int lane) {
  int row = lane & 15;
  int kb  = (lane >> 4) * 8;
  union { uint4 u[2]; v16bf v; } f;
  const char* base = (const char*)(p + row * stride + kb);
  f.u[0] = *(const uint4*)(base);
  f.u[1] = *(const uint4*)(base + 32);   // +16 elements
  return f.v;
}

// B-matrix 32x16 bf16 fragment; staged tile holds column n of B as row n.
// Lane L<16: col=L, K={0..7, 16..23}; lane L>=16: col=L-16, K={8..15, 24..31}.
__device__ __forceinline__ v16bf frag_b(const bf16* p, int stride, int lane) {
  int col = lane & 15;
  int kb  = (lane >> 4) * 8;
  union { uint4 u[2]; v16bf v; } f;
  const char* base = (const char*)(p + col * stride + kb);
  f.u[0] = *(const uint4*)(base);
  f.u[1] = *(const uint4*)(base + 32);   // +16 elements
  return f.v;
}

// half-wave (16-lane) reductions: rows of a 16x16 C tile live in one half-wave
__device__ __forceinline__ float halfmax(float v) {
#pragma unroll
  for (int m = 1; m <= 8; m <<= 1) v = fmaxf(v, __shfl_xor(v, m, 32));
  return v;
}
__device__ __forceinline__ float halfsum(float v) {
#pragma unroll
  for (int m = 1; m <= 8; m <<= 1) v += __shfl_xor(v, m, 32);
  return v;
}

// ---------------- fp32 -> bf16 conversion ----------------------------------

__global__ __launch_bounds__(256) void cvt_f32_bf16(const float* __restrict__ in,
                                                    bf16* __restrict__ out, int n) {
  int i = (blockIdx.x * blockDim.x + threadIdx.x) * 4;
  int stride = gridDim.x * blockDim.x * 4;
  for (; i < n; i += stride) {
    float4 f = *(const float4*)(in + i);
    union { bf16 h[4]; uint2 u; } o;
    o.h[0] = (bf16)f.x; o.h[1] = (bf16)f.y;
    o.h[2] = (bf16)f.z; o.h[3] = (bf16)f.w;
    *(uint2*)(out + i) = o.u;
  }
}

// ---------------- GEMM: C[M,N] = A[M,K] * W[N,K]^T  (bf16 in, f32 acc) -----
// 128x128 block, BK=64, double-buffered async-to-LDS staging.

#define BM 128
#define BN 128
#define BK 64
#define LDP 72   // padded LDS stride (16B aligned, breaks bank stride)

template <int OUT_BF16>
__global__ __launch_bounds__(256)
void gemm_bf16(const bf16* __restrict__ A, const bf16* __restrict__ W,
               float* __restrict__ Cf, bf16* __restrict__ Cb,
               int M, int N, int K) {
  __shared__ bf16 at[2][BM][LDP];
  __shared__ bf16 wt[2][BN][LDP];

  const int tid  = threadIdx.x;
  const int lane = tid & 31;
  const int wave = tid >> 5;
  const int wm   = (wave >> 1) * 32;   // 4 waves down M
  const int wn   = (wave & 1) * 64;    // 2 waves across N
  const int m0   = blockIdx.y * BM;
  const int n0   = blockIdx.x * BN;

  v8f acc[2][4];
#pragma unroll
  for (int i = 0; i < 2; ++i)
#pragma unroll
    for (int j = 0; j < 4; ++j) acc[i][j] = (v8f){0.f,0.f,0.f,0.f,0.f,0.f,0.f,0.f};

  // prologue: async-stage first K-tile into buffer 0
#pragma unroll
  for (int idx = tid; idx < BM * (BK / 8); idx += 256) {
    int r = idx >> 3, c = (idx & 7) * 8;
    async_ld16(&at[0][r][c], A + (long)(m0 + r) * K + c);
    async_ld16(&wt[0][r][c], W + (long)(n0 + r) * K + c);
  }
  wait_async();
  __syncthreads();

  int buf = 0;
  for (int k0 = 0; k0 < K; k0 += BK, buf ^= 1) {
    const int kn = k0 + BK;
    if (kn < K) {  // async-stage next K-tile into the other buffer
#pragma unroll
      for (int idx = tid; idx < BM * (BK / 8); idx += 256) {
        int r = idx >> 3, c = (idx & 7) * 8;
        async_ld16(&at[buf ^ 1][r][c], A + (long)(m0 + r) * K + kn + c);
        async_ld16(&wt[buf ^ 1][r][c], W + (long)(n0 + r) * K + kn + c);
      }
    }

    // compute on current buffer: two k=32 WMMA steps
#pragma unroll
    for (int ks = 0; ks < BK; ks += 32) {
      v16bf af[2];
#pragma unroll
      for (int mt = 0; mt < 2; ++mt)
        af[mt] = frag_a(&at[buf][wm + mt * 16][ks], LDP, lane);
#pragma unroll
      for (int nt = 0; nt < 4; ++nt) {
        v16bf bfr = frag_b(&wt[buf][wn + nt * 16][ks], LDP, lane);
#pragma unroll
        for (int mt = 0; mt < 2; ++mt)
          acc[mt][nt] = wmma_bf16(af[mt], bfr, acc[mt][nt]);
      }
    }
    wait_async();     // own async loads into buf^1 done
    __syncthreads();  // everyone's loads done, everyone finished reading buf
  }

  // epilogue: C layout lane L, VGPR r -> M = r + 8*(L>=16), N = L&15
  const int crow = (lane >> 4) * 8;
  const int ccol = lane & 15;
#pragma unroll
  for (int mt = 0; mt < 2; ++mt)
#pragma unroll
    for (int nt = 0; nt < 4; ++nt)
#pragma unroll
      for (int r = 0; r < 8; ++r) {
        long m = m0 + wm + mt * 16 + crow + r;
        long n = n0 + wn + nt * 16 + ccol;
        float v = acc[mt][nt][r];
        if (OUT_BF16) Cb[m * N + n] = (bf16)v;
        else          Cf[m * N + n] = v;
      }
}

// ---------------- RoPE (in-place on bf16 Q and K) --------------------------

__global__ __launch_bounds__(256) void rope_inplace(bf16* __restrict__ q,
                                                    bf16* __restrict__ k) {
  long idx = (long)blockIdx.x * blockDim.x + threadIdx.x;  // (bs, h, d<32)
  int  d   = (int)(idx & 31);
  long t1  = idx >> 5;
  int  h   = (int)(t1 & (H_ - 1));
  long bs  = t1 >> 5;                      // H_ == 32
  int  s   = (int)(bs & (S_ - 1));
  long base = bs * D_ + (long)h * HD_;

  float inv = __powf(10000.f, -(float)d * (1.f / 32.f));
  float c, sn;
  __sincosf((float)s * inv, &sn, &c);

  float q1 = (float)q[base + d], q2 = (float)q[base + d + 32];
  q[base + d]      = (bf16)(q1 * c - q2 * sn);
  q[base + d + 32] = (bf16)(q2 * c + q1 * sn);
  float k1 = (float)k[base + d], k2 = (float)k[base + d + 32];
  k[base + d]      = (bf16)(k1 * c - k2 * sn);
  k[base + d + 32] = (bf16)(k2 * c + k1 * sn);
}

// ---------------- Flash attention with window + global-token mask ----------
// grid: (S/64, B*H). 4 waves; each wave owns 16 query rows.

#define LQ 72  // padded LDS stride

__global__ __launch_bounds__(128)
void flash_attn(const bf16* __restrict__ qb, const bf16* __restrict__ kb,
                const bf16* __restrict__ vb, const unsigned char* __restrict__ gm,
                bf16* __restrict__ ob) {
  __shared__ bf16  qt[64][LQ];
  __shared__ bf16  kt[64][LQ];
  __shared__ bf16  vt[64][LQ];        // V transposed: vt[d][j]
  __shared__ bf16  pst[4][16][LQ];    // per-wave P staging
  __shared__ float growf[64], gcolf[64];

  const int tid = threadIdx.x, lane = tid & 31, wave = tid >> 5;
  const int bh = blockIdx.y, b = bh >> 5, h = bh & 31;
  const int i0 = blockIdx.x * 64;
  const long hb = ((long)b * S_) * D_ + (long)h * HD_;  // GQA = flat channel slice

  // async-stage Q tile; row global-flags
  for (int idx = tid; idx < 64 * 8; idx += 128) {
    int r = idx >> 3, c = (idx & 7) * 8;
    async_ld16(&qt[r][c], qb + hb + (long)(i0 + r) * D_ + c);
  }
  for (int r = tid; r < 64; r += 128) growf[r] = gm[b * S_ + i0 + r] ? 1.f : 0.f;

  v8f oacc[4];
#pragma unroll
  for (int dt = 0; dt < 4; ++dt) oacc[dt] = (v8f){0.f,0.f,0.f,0.f,0.f,0.f,0.f,0.f};
  float mrow[8], lrow[8];
#pragma unroll
  for (int r = 0; r < 8; ++r) { mrow[r] = NEGF; lrow[r] = 0.f; }

  const int half = lane >> 4, cc = lane & 15;
  wait_async();
  __syncthreads();

  for (int j0 = 0; j0 < S_; j0 += 64) {
    // K rows: async-to-LDS (contiguous B-fragments for QK^T)
    for (int idx = tid; idx < 64 * 8; idx += 128) {
      int r = idx >> 3, c = (idx & 7) * 8;
      async_ld16(&kt[r][c], kb + hb + (long)(j0 + r) * D_ + c);
    }
    // V transposed (synchronous scatter; overlaps the in-flight K asyncs)
    for (int idx = tid; idx < 64 * 8; idx += 128) {
      int r = idx >> 3, c = (idx & 7) * 8;
      union { uint4 u; bf16 hh[8]; } vv;
      vv.u = *(const uint4*)(vb + hb + (long)(j0 + r) * D_ + c);
#pragma unroll
      for (int e = 0; e < 8; ++e) vt[c + e][r] = vv.hh[e];
    }
    for (int r = tid; r < 64; r += 128) gcolf[r] = gm[b * S_ + j0 + r] ? 1.f : 0.f;
    wait_async();
    __syncthreads();

    // S = Q K^T  (16x64 per wave, two k=32 steps)
    v8f sacc[4];
#pragma unroll
    for (int nt = 0; nt < 4; ++nt) sacc[nt] = (v8f){0.f,0.f,0.f,0.f,0.f,0.f,0.f,0.f};
#pragma unroll
    for (int ks = 0; ks < 64; ks += 32) {
      v16bf aq = frag_a(&qt[wave * 16][ks], LQ, lane);
#pragma unroll
      for (int nt = 0; nt < 4; ++nt) {
        v16bf bk = frag_b(&kt[nt * 16][ks], LQ, lane);
        sacc[nt] = wmma_bf16(aq, bk, sacc[nt]);
      }
    }

    // scale + sliding-window / global-token mask
#pragma unroll
    for (int nt = 0; nt < 4; ++nt) {
      int   j  = j0 + nt * 16 + cc;
      float gc = gcolf[nt * 16 + cc];
#pragma unroll
      for (int r = 0; r < 8; ++r) {
        int   i    = i0 + wave * 16 + half * 8 + r;
        float gr   = growf[wave * 16 + half * 8 + r];
        int   diff = i - j;
        bool  inw  = (diff >= 0) && (diff < WIN_);
        float mv   = (inw || gr > 0.5f || gc > 0.5f) ? 0.f : NEGF;
        sacc[nt][r] = sacc[nt][r] * 0.125f + mv;   // 1/sqrt(64)
      }
    }

    // online softmax (rows live in a 16-lane half-wave)
    float mnew[8], alpha[8];
#pragma unroll
    for (int r = 0; r < 8; ++r) {
      float t = fmaxf(fmaxf(sacc[0][r], sacc[1][r]), fmaxf(sacc[2][r], sacc[3][r]));
      t        = halfmax(t);
      mnew[r]  = fmaxf(mrow[r], t);
      alpha[r] = __expf(mrow[r] - mnew[r]);
      mrow[r]  = mnew[r];
    }
#pragma unroll
    for (int nt = 0; nt < 4; ++nt)
#pragma unroll
      for (int r = 0; r < 8; ++r) sacc[nt][r] = __expf(sacc[nt][r] - mnew[r]);
#pragma unroll
    for (int r = 0; r < 8; ++r) {
      float s = sacc[0][r] + sacc[1][r] + sacc[2][r] + sacc[3][r];
      lrow[r] = lrow[r] * alpha[r] + halfsum(s);
    }
#pragma unroll
    for (int dt = 0; dt < 4; ++dt)
#pragma unroll
      for (int r = 0; r < 8; ++r) oacc[dt][r] *= alpha[r];

    // stage P (per-wave region; DS ops are in-order within a wave)
#pragma unroll
    for (int nt = 0; nt < 4; ++nt)
#pragma unroll
      for (int r = 0; r < 8; ++r)
        pst[wave][half * 8 + r][nt * 16 + cc] = (bf16)sacc[nt][r];

    // O += P @ V
#pragma unroll
    for (int ks = 0; ks < 64; ks += 32) {
      v16bf ap = frag_a(&pst[wave][0][ks], LQ, lane);
#pragma unroll
      for (int dt = 0; dt < 4; ++dt) {
        v16bf bv = frag_b(&vt[dt * 16][ks], LQ, lane);
        oacc[dt] = wmma_bf16(ap, bv, oacc[dt]);
      }
    }
    __syncthreads();   // before kt/vt are restaged
  }

  // normalize + write bf16 output tile
#pragma unroll
  for (int r = 0; r < 8; ++r) {
    float invl = 1.f / lrow[r];
    int   i    = i0 + wave * 16 + half * 8 + r;
#pragma unroll
    for (int dt = 0; dt < 4; ++dt)
      ob[hb + (long)i * D_ + dt * 16 + cc] = (bf16)(oacc[dt][r] * invl);
  }
}

// ---------------- launcher -------------------------------------------------

extern "C" void kernel_launch(void* const* d_in, const int* in_sizes, int n_in,
                              void* d_out, int out_size, void* d_ws, size_t ws_size,
                              hipStream_t stream) {
  (void)in_sizes; (void)n_in; (void)out_size; (void)ws_size;
  const float* x  = (const float*)d_in[0];
  const float* Wq = (const float*)d_in[1];
  const float* Wk = (const float*)d_in[2];
  const float* Wv = (const float*)d_in[3];
  const float* Wo = (const float*)d_in[4];
  const unsigned char* gmask = (const unsigned char*)d_in[5];

  const int M = B_ * S_, N = D_, K = D_;
  char* ws = (char*)d_ws;
  size_t off = 0;
  bf16* xb  = (bf16*)(ws + off); off += (size_t)M * D_ * 2;
  bf16* wqb = (bf16*)(ws + off); off += (size_t)D_ * D_ * 2;
  bf16* wkb = (bf16*)(ws + off); off += (size_t)D_ * D_ * 2;
  bf16* wvb = (bf16*)(ws + off); off += (size_t)D_ * D_ * 2;
  bf16* wob = (bf16*)(ws + off); off += (size_t)D_ * D_ * 2;
  bf16* qbu = (bf16*)(ws + off); off += (size_t)M * D_ * 2;
  bf16* kbu = (bf16*)(ws + off); off += (size_t)M * D_ * 2;
  bf16* vbu = (bf16*)(ws + off); off += (size_t)M * D_ * 2;
  bf16* att = (bf16*)(ws + off); off += (size_t)M * D_ * 2;

  // 1) bf16 conversions
  cvt_f32_bf16<<<(M * D_) / 1024, 256, 0, stream>>>(x, xb, M * D_);
  cvt_f32_bf16<<<(D_ * D_) / 1024, 256, 0, stream>>>(Wq, wqb, D_ * D_);
  cvt_f32_bf16<<<(D_ * D_) / 1024, 256, 0, stream>>>(Wk, wkb, D_ * D_);
  cvt_f32_bf16<<<(D_ * D_) / 1024, 256, 0, stream>>>(Wv, wvb, D_ * D_);
  cvt_f32_bf16<<<(D_ * D_) / 1024, 256, 0, stream>>>(Wo, wob, D_ * D_);

  // 2) Q/K/V projections (bf16 out)
  dim3 gg(N / BN, M / BM);
  gemm_bf16<1><<<gg, 256, 0, stream>>>(xb, wqb, nullptr, qbu, M, N, K);
  gemm_bf16<1><<<gg, 256, 0, stream>>>(xb, wkb, nullptr, kbu, M, N, K);
  gemm_bf16<1><<<gg, 256, 0, stream>>>(xb, wvb, nullptr, vbu, M, N, K);

  // 3) RoPE in place on Q and K
  rope_inplace<<<(B_ * S_ * H_ * 32) / 256, 256, 0, stream>>>(qbu, kbu);

  // 4) attention (GQA is the identity in flat channel space)
  dim3 ga(S_ / 64, B_ * H_);
  flash_attn<<<ga, 128, 0, stream>>>(qbu, kbu, vbu, gmask, att);

  // 5) output projection (fp32 out)
  gemm_bf16<0><<<gg, 256, 0, stream>>>(att, wob, (float*)d_out, nullptr, M, N, K);
}
